// Attention_50964081934424
// MI455X (gfx1250) — matrix operand order repaired and checked by reading, not compiled, not run on hardware
//
#include <hip/hip_runtime.h>

#define LNQ 4096
#define IDM 512
#define HDM 64
#define NB 4

typedef _Float16 half8 __attribute__((ext_vector_type(8)));
typedef _Float16 v16h  __attribute__((ext_vector_type(16)));
typedef float    v8f   __attribute__((ext_vector_type(8)));
typedef float    f32x8 __attribute__((ext_vector_type(8)));

__device__ __forceinline__ v16h mk16(half8 lo, half8 hp) {
  return __builtin_shufflevector(lo, hp, 0, 1, 2, 3, 4, 5, 6, 7, 8, 9, 10, 11, 12, 13, 14, 15);
}

__device__ __forceinline__ v8f wmma16(v16h a, v16h b, v8f c) {
  return __builtin_amdgcn_wmma_f32_16x16x32_f16(false, a, false, b, (short)0, c, false, false);
}

// Reductions across each 16-lane half of the wave (xor masks < 16 never cross
// the bit-4 boundary, matching the C-fragment row split).
__device__ __forceinline__ float rmax16(float v) {
  v = fmaxf(v, __shfl_xor(v, 1, 32));
  v = fmaxf(v, __shfl_xor(v, 2, 32));
  v = fmaxf(v, __shfl_xor(v, 4, 32));
  v = fmaxf(v, __shfl_xor(v, 8, 32));
  return v;
}
__device__ __forceinline__ float rsum16(float v) {
  v += __shfl_xor(v, 1, 32);
  v += __shfl_xor(v, 2, 32);
  v += __shfl_xor(v, 4, 32);
  v += __shfl_xor(v, 8, 32);
  return v;
}

// Load the 4 K^T B-fragments (two 16-key subtiles x two 32-dim halves).
__device__ __forceinline__ void loadKfrag(const _Float16* __restrict__ Kbase,
                                          int kb, int ln15, int hi,
                                          v16h* dst) {
#pragma unroll
  for (int sub = 0; sub < 2; ++sub) {
    const _Float16* krow = Kbase + (size_t)(kb + 16 * sub + ln15) * HDM;
#pragma unroll
    for (int f = 0; f < 2; ++f) {
      const _Float16* bp = krow + 32 * f + hi * 8;
      dst[sub * 2 + f] = mk16(*(const half8*)bp, *(const half8*)(bp + 16));
    }
  }
}

// ---------------------------------------------------------------------------
// Phase 0: transpose weights to f16 so WMMA B-operand loads are contiguous.
//   qt/kt/v1t : [HDM][IDM]   (from [IDM][HDM] f32)
//   v2t       : [IDM][HDM]   (from [HDM][IDM] f32)
// ---------------------------------------------------------------------------
__global__ __launch_bounds__(256) void prep_weights(
    const float* __restrict__ kw, const float* __restrict__ qw,
    const float* __restrict__ v1w, const float* __restrict__ v2w,
    _Float16* __restrict__ qt, _Float16* __restrict__ kt,
    _Float16* __restrict__ v1t, _Float16* __restrict__ v2t) {
  int idx = blockIdx.x * blockDim.x + threadIdx.x;  // 4 * 32768 threads exactly
  int m = idx >> 15;
  int r = idx & 32767;
  if (m == 0) {
    int n = r / IDM, kd = r % IDM;
    qt[n * IDM + kd] = (_Float16)qw[kd * HDM + n];
  } else if (m == 1) {
    int n = r / IDM, kd = r % IDM;
    kt[n * IDM + kd] = (_Float16)kw[kd * HDM + n];
  } else if (m == 2) {
    int n = r / IDM, kd = r % IDM;
    v1t[n * IDM + kd] = (_Float16)v1w[kd * HDM + n];
  } else {
    int n = r / HDM, kd = r % HDM;
    v2t[n * HDM + kd] = (_Float16)v2w[kd * IDM + n];
  }
}

// ---------------------------------------------------------------------------
// Phase 1: projections. Each wave computes a 16-row tile of Q = i@q, K = i@k,
// P = i@v1 via WMMA in a single k-loop (one A-fragment feeds 12 WMMAs).
// B-fragments batched per weight matrix (8 b128 loads in flight) for MLP.
// Q,K stored row-major f16; P stored transposed [B][HDM][LN].
// ---------------------------------------------------------------------------
__global__ __launch_bounds__(128) void proj_kernel(
    const float* __restrict__ inp, const _Float16* __restrict__ qt,
    const _Float16* __restrict__ kt, const _Float16* __restrict__ v1t,
    _Float16* __restrict__ Qf, _Float16* __restrict__ Kf,
    _Float16* __restrict__ Pt) {
  const int lane = threadIdx.x & 31;
  const int wv = threadIdx.x >> 5;
  const int hi = lane >> 4;
  const int ln15 = lane & 15;
  const int tile = blockIdx.x * 4 + wv;  // 0..1023
  const int rowbase = tile * 16;         // flattened row in [0, B*LN)
  const int batch = rowbase / LNQ;
  const int pos = rowbase % LNQ;

  const float* irow = inp + (size_t)(rowbase + ln15) * IDM;
  const _Float16* Ws[3] = {qt, kt, v1t};

  v8f acc[12] = {};
  for (int ks = 0; ks < IDM / 32; ++ks) {
    const int kb = ks * 32;
    const int c0 = kb + hi * 8;
    // A fragment (16x32 f16): per lane two 8-element K chunks, cvt f32->f16
    half8 lo = __builtin_convertvector(*(const f32x8*)(irow + c0), half8);
    half8 hp = __builtin_convertvector(*(const f32x8*)(irow + c0 + 16), half8);
    v16h A = mk16(lo, hp);
#pragma unroll
    for (int w = 0; w < 3; ++w) {
      // batch the 4 B-fragment loads (one clause, 8 loads in flight) ...
      v16h Bf[4];
#pragma unroll
      for (int t = 0; t < 4; ++t) {
        const _Float16* bp =
            Ws[w] + (size_t)(16 * t + ln15) * IDM + kb + hi * 8;
        Bf[t] = mk16(*(const half8*)bp, *(const half8*)(bp + 16));
      }
      // ... then issue the 4 WMMAs with staggered load waits
#pragma unroll
      for (int t = 0; t < 4; ++t)
        acc[w * 4 + t] = wmma16(A, Bf[t], acc[w * 4 + t]);
    }
  }
  // Q, K: row-major f16; P: transposed
#pragma unroll
  for (int t = 0; t < 4; ++t)
#pragma unroll
    for (int r = 0; r < 8; ++r) {
      Qf[(size_t)(rowbase + r + 8 * hi) * HDM + 16 * t + ln15] =
          (_Float16)acc[t][r];
      Kf[(size_t)(rowbase + r + 8 * hi) * HDM + 16 * t + ln15] =
          (_Float16)acc[4 + t][r];
      Pt[(size_t)(batch * HDM + 16 * t + ln15) * LNQ + pos + r + 8 * hi] =
          (_Float16)acc[8 + t][r];
    }
}

// ---------------------------------------------------------------------------
// Phase 2: causal flash attention + fused final @v2.
// One wave per 16-query tile; tiles paired (j, 255-j) per block for causal
// load balance. Online softmax (base-2) fully in C-fragment registers via
// shfl_xor. Software-pipelined: next chunk's K fragments and this chunk's P
// fragments are loaded BEFORE the softmax + LDS fence so global latency is
// hidden behind the softmax VALU work.
// ---------------------------------------------------------------------------
__global__ __launch_bounds__(128) void flash_kernel(
    const _Float16* __restrict__ Qf, const _Float16* __restrict__ Kf,
    const _Float16* __restrict__ Pt, const _Float16* __restrict__ v2t,
    float* __restrict__ out) {
  __shared__ __attribute__((aligned(32))) _Float16 sP[4][16 * 32];
  __shared__ __attribute__((aligned(32))) _Float16 sO[4][16 * 64];

  const int lane = threadIdx.x & 31;
  const int wv = threadIdx.x >> 5;
  const int hi = lane >> 4;
  const int ln15 = lane & 15;

  // balanced tile assignment: pair short tile j with long tile 255-j
  const int g = blockIdx.x * 4 + wv;  // 0..1023
  const int batch = g >> 8;
  const int u = g & 255;
  const int tib = (u & 1) ? (255 - (u >> 1)) : (u >> 1);
  const int qb = tib * 16;

  // Q A-fragments (16 queries x 64 dims) held in registers
  const _Float16* qrow = Qf + (size_t)(batch * LNQ + qb + ln15) * HDM;
  v16h Qa[2];
#pragma unroll
  for (int f = 0; f < 2; ++f) {
    const _Float16* p = qrow + 32 * f + hi * 8;
    Qa[f] = mk16(*(const half8*)p, *(const half8*)(p + 16));
  }

  v8f O[4] = {};
  float mrow[8], lrow[8];
#pragma unroll
  for (int r = 0; r < 8; ++r) {
    mrow[r] = -__builtin_inff();
    lrow[r] = 0.0f;
  }

  const _Float16* Kbase = Kf + (size_t)(batch * LNQ) * HDM;
  const _Float16* Pbase = Pt + (size_t)(batch * HDM) * LNQ;

  // scale 1/sqrt(64) folded with log2(e): softmax done in base 2
  const float SC = 0.125f * 1.44269504088896340736f;

  v16h Kcur[4], Knxt[4];
  loadKfrag(Kbase, 0, ln15, hi, Kcur);

  const int kend = qb + 15;
  for (int kb = 0; kb <= kend; kb += 32) {
    // ---- S = Q @ K^T for this 16x32 chunk (4 WMMAs, K frags preloaded) ----
    v8f Sc[2];
#pragma unroll
    for (int sub = 0; sub < 2; ++sub) {
      v8f s = {};
      s = wmma16(Qa[0], Kcur[sub * 2 + 0], s);
      s = wmma16(Qa[1], Kcur[sub * 2 + 1], s);
      Sc[sub] = s;
    }

    // ---- issue independent loads BEFORE the softmax/LDS fence ----
    // P B-fragments for this chunk (consumed after softmax)
    v16h Pb[4];
#pragma unroll
    for (int t = 0; t < 4; ++t) {
      const _Float16* bp = Pbase + (size_t)(16 * t + ln15) * LNQ + kb + hi * 8;
      Pb[t] = mk16(*(const half8*)bp, *(const half8*)(bp + 16));
    }
    // next chunk's K fragments (harmless overrun into adjacent ws region)
    loadKfrag(Kbase, kb + 32, ln15, hi, Knxt);

    // ---- online softmax in C-fragment layout (all 32 lanes active) ----
    const int j0 = kb + ln15;
    const int j1 = kb + 16 + ln15;
#pragma unroll
    for (int r = 0; r < 8; ++r) {
      const int qr = qb + r + 8 * hi;
      float s0 = (j0 <= qr) ? Sc[0][r] * SC : -__builtin_inff();
      float s1 = (j1 <= qr) ? Sc[1][r] * SC : -__builtin_inff();
      float mx = rmax16(fmaxf(s0, s1));
      mx = fmaxf(mx, mrow[r]);
      float alpha = __builtin_amdgcn_exp2f(mrow[r] - mx);
      float p0 = __builtin_amdgcn_exp2f(s0 - mx);
      float p1 = __builtin_amdgcn_exp2f(s1 - mx);
      lrow[r] = lrow[r] * alpha + rsum16(p0 + p1);
      mrow[r] = mx;
      O[0][r] *= alpha;
      O[1][r] *= alpha;
      O[2][r] *= alpha;
      O[3][r] *= alpha;
      sP[wv][(r + 8 * hi) * 32 + ln15] = (_Float16)p0;
      sP[wv][(r + 8 * hi) * 32 + 16 + ln15] = (_Float16)p1;
    }
    asm volatile("s_wait_dscnt 0x0" ::: "memory");

    // ---- probs A-fragment from LDS, O += P @ PV (4 WMMAs) ----
    const _Float16* pp = &sP[wv][ln15 * 32 + hi * 8];
    v16h Pa = mk16(*(const half8*)pp, *(const half8*)(pp + 16));
#pragma unroll
    for (int t = 0; t < 4; ++t) O[t] = wmma16(Pa, Pb[t], O[t]);

#pragma unroll
    for (int z = 0; z < 4; ++z) Kcur[z] = Knxt[z];
  }

  // ---- epilogue: normalize, transpose O via LDS, fused @v2 ----
  float linv[8];
#pragma unroll
  for (int r = 0; r < 8; ++r) linv[r] = 1.0f / lrow[r];
#pragma unroll
  for (int t = 0; t < 4; ++t)
#pragma unroll
    for (int r = 0; r < 8; ++r)
      sO[wv][(r + 8 * hi) * 64 + 16 * t + ln15] =
          (_Float16)(O[t][r] * linv[r]);
  asm volatile("s_wait_dscnt 0x0" ::: "memory");

  v16h Oa[2];
#pragma unroll
  for (int f = 0; f < 2; ++f) {
    const _Float16* p = &sO[wv][ln15 * 64 + 32 * f + hi * 8];
    Oa[f] = mk16(*(const half8*)p, *(const half8*)(p + 16));
  }

  float* orow = out + (size_t)(batch * LNQ + qb) * IDM;
#pragma unroll 2
  for (int t = 0; t < 32; ++t) {
    v8f acc = {};
#pragma unroll
    for (int f = 0; f < 2; ++f) {
      const _Float16* bp = v2t + (size_t)(16 * t + ln15) * HDM + 32 * f + hi * 8;
      v16h Bf = mk16(*(const half8*)bp, *(const half8*)(bp + 16));
      acc = wmma16(Oa[f], Bf, acc);
    }
#pragma unroll
    for (int r = 0; r < 8; ++r)
      orow[(size_t)(r + 8 * hi) * IDM + 16 * t + ln15] = acc[r];
  }
}

// ---------------------------------------------------------------------------
// Launch: prep -> proj -> flash on the same stream (implicit ordering).
// Workspace layout (f16): qt, kt, v1t [64*512] each, v2t [512*64],
// Qf, Kf [B*LN*64], Pt [B*64*LN]  -> ~6.6 MB total.
// ---------------------------------------------------------------------------
extern "C" void kernel_launch(void* const* d_in, const int* in_sizes, int n_in,
                              void* d_out, int out_size, void* d_ws,
                              size_t ws_size, hipStream_t stream) {
  (void)in_sizes; (void)n_in; (void)out_size; (void)ws_size;
  const float* i_ = (const float*)d_in[0];
  const float* kw = (const float*)d_in[1];
  const float* qw = (const float*)d_in[2];
  const float* v1 = (const float*)d_in[3];
  const float* v2 = (const float*)d_in[4];
  float* out = (float*)d_out;

  _Float16* wsh = (_Float16*)d_ws;
  _Float16* qt = wsh;
  _Float16* kt = qt + (size_t)HDM * IDM;
  _Float16* v1t = kt + (size_t)HDM * IDM;
  _Float16* v2t = v1t + (size_t)HDM * IDM;
  _Float16* Qf = v2t + (size_t)IDM * HDM;
  _Float16* Kf = Qf + (size_t)NB * LNQ * HDM;
  _Float16* Pt = Kf + (size_t)NB * LNQ * HDM;

  prep_weights<<<512, 256, 0, stream>>>(kw, qw, v1, v2, qt, kt, v1t, v2t);
  proj_kernel<<<256, 128, 0, stream>>>(i_, qt, kt, v1t, Qf, Kf, Pt);
  flash_kernel<<<256, 128, 0, stream>>>(Qf, Kf, Pt, v2t, out);
}
